// HieroLMAttention_61272003444956
// MI455X (gfx1250) — matrix operand order repaired and checked by reading, not compiled, not run on hardware
//
#include <hip/hip_runtime.h>
#include <hip/hip_bf16.h>
#include <math.h>

// ---------------- problem dims ----------------
#define TT 256
#define BB 8
#define HH 256
#define EE 256
#define VV 32000
#define NEGF (-1.0e10f)

// ---------------- WMMA / TDM types ----------------
typedef __attribute__((ext_vector_type(16))) __bf16       v16bf;
typedef __attribute__((ext_vector_type(8)))  float        v8f;
typedef __attribute__((ext_vector_type(2)))  float        v2f;
typedef __attribute__((ext_vector_type(8)))  unsigned int v8u;
typedef __attribute__((ext_vector_type(4)))  unsigned int v4u;
typedef __attribute__((ext_vector_type(8)))  int          v8i;
typedef __attribute__((ext_vector_type(4)))  int          v4i;

static __device__ __forceinline__ v8f wmma_bf16(v16bf a, v16bf b, v8f c) {
  return __builtin_amdgcn_wmma_f32_16x16x32_bf16(false, a, false, b, (short)0, c, false, false);
}
static __device__ __forceinline__ v8f wmma_f32x4(v2f a, v2f b, v8f c) {
  return __builtin_amdgcn_wmma_f32_16x16x4_f32(false, a, false, b, (short)0, c, false, false);
}

// A operand: 16xK row-major bf16 tile (A points at row 0 of the tile).
// Layout (contiguous K order, matches B loaders): lane<16 -> K 0..15, lane>=16 -> K 16..31.
static __device__ __forceinline__ v16bf load_a_bf16(const __hip_bfloat16* A, int lda,
                                                    int k0, int lane) {
  int row = lane & 15, half = lane >> 4;
  const unsigned short* p = (const unsigned short*)(A + (size_t)row * lda);
  v8u u;
#pragma unroll
  for (int i = 0; i < 4; ++i)
    u[i] = *(const unsigned int*)(p + k0 + 16 * half + 2 * i);
#pragma unroll
  for (int i = 0; i < 4; ++i)
    u[4 + i] = *(const unsigned int*)(p + k0 + 16 * half + 8 + 2 * i);
  return __builtin_bit_cast(v16bf, u);
}

// B operand from W^T stored row-major (N x K): per lane column n = n0+lane&15,
// consecutive K contiguous.  Wt must point at row n0.
static __device__ __forceinline__ v16bf load_b_wt(const __hip_bfloat16* Wt, int ldw,
                                                  int k0, int lane) {
  int col = lane & 15, half = lane >> 4;
  const unsigned short* p = (const unsigned short*)(Wt + (size_t)col * ldw);
  v8u u;
#pragma unroll
  for (int i = 0; i < 8; ++i)
    u[i] = *(const unsigned int*)(p + k0 + 16 * half + 2 * i);
  return __builtin_bit_cast(v16bf, u);
}

// B operand from a K x N row-major matrix with row stride ldb (strided 16-bit loads).
static __device__ __forceinline__ v16bf load_b_strided(const __hip_bfloat16* B, int ldb,
                                                       int k0, int lane) {
  int col = lane & 15, half = lane >> 4;
  const unsigned short* p = (const unsigned short*)B;
  v8u u;
#pragma unroll
  for (int i = 0; i < 8; ++i) {
    int k = k0 + 16 * half + 2 * i;
    unsigned int lo = p[(size_t)k * ldb + col];
    unsigned int hi = p[(size_t)(k + 1) * ldb + col];
    u[i] = lo | (hi << 16);
  }
  return __builtin_bit_cast(v16bf, u);
}

static __device__ __forceinline__ float sigf(float x) { return 1.f / (1.f + expf(-x)); }

// ---------------- TDM: load a contiguous (rows x cols) bf16 tile into LDS ----------------
// D# packing per CDNA5 ISA §8.3/8.4: group0 = {count=1, lds_addr, global_addr, type=2},
// group1 = {data_size=2B, tensor_dim0=cols, tensor_dim1=rows, tile_dim0=cols,
//           tile_dim1=rows, tensor_dim0_stride=cols}.  2D -> groups 2/3 zero.
#if __has_builtin(__builtin_amdgcn_tensor_load_to_lds) && \
    __has_builtin(__builtin_amdgcn_s_wait_tensorcnt)
#define HAVE_TDM 1
static __device__ __forceinline__ void tdm_load_bf16_tile(unsigned lds_off,
                                                          unsigned long long gaddr,
                                                          unsigned cols, unsigned rows) {
  v4u g0 = {};
  g0[0] = 1u;                                        // count=1, user descriptor
  g0[1] = lds_off;                                   // LDS byte address
  g0[2] = (unsigned)gaddr;                           // global_addr[31:0]  (bits 95:64)
  g0[3] = ((unsigned)(gaddr >> 32) & 0x01FFFFFFu)    // global_addr[56:32]
          | (2u << 30);                              // type = 2 ("image")
  v8i g1 = {};
  g1[0] = (int)(1u << 16);                           // data_size = 1 -> 2-byte elements
  g1[1] = (int)((cols & 0xFFFFu) << 16);             // tensor_dim0[15:0]
  g1[2] = (int)((cols >> 16) | ((rows & 0xFFFFu) << 16));  // td0 hi | tensor_dim1 lo
  g1[3] = (int)((rows >> 16) | (cols << 16));        // td1 hi | tile_dim0
  g1[4] = (int)rows;                                 // tile_dim1 (tile_dim2 = 0)
  g1[5] = (int)cols;                                 // tensor_dim0_stride[31:0]
  g1[6] = 0; g1[7] = 0;
  v4i gz = {};
#if __clang_major__ >= 23
  v8i gz8 = {};
  __builtin_amdgcn_tensor_load_to_lds(g0, g1, gz, gz, gz8, 0);
#else
  __builtin_amdgcn_tensor_load_to_lds(g0, g1, gz, gz, 0);
#endif
}
#else
#define HAVE_TDM 0
#endif

// ---------------- small utility kernels ----------------
__global__ void zero_kernel(float* out, int n) {
  int i = blockIdx.x * blockDim.x + threadIdx.x;
  if (i < n) out[i] = 0.f;
}

__global__ void cast_bf16_kernel(const float* in, __hip_bfloat16* out, int n) {
  int i = blockIdx.x * blockDim.x + threadIdx.x;
  int stride = gridDim.x * blockDim.x;
  for (; i < n; i += stride) out[i] = __float2bfloat16(in[i]);
}

// gather + cast: x[t,b,:] = emb[source[t,b]]
__global__ void embed_kernel(const int* source, const float* emb, __hip_bfloat16* xbf) {
  int row = blockIdx.x, e = threadIdx.x;
  int idx = source[row];
  xbf[(size_t)row * EE + e] = __float2bfloat16(emb[(size_t)idx * EE + e]);
}

// ---------------- stage B: xg = x @ w_ih^T + b_ih + b_hh  (2048x1024) ----------------
__global__ void xw_gemm_kernel(const __hip_bfloat16* xbf, const __hip_bfloat16* wih_bf,
                               const float* b_ih, const float* b_hh, float* xg) {
  int lane = threadIdx.x & 31, wave = threadIdx.x >> 5;
  int tile = blockIdx.x * 4 + wave;           // 8192 tiles: 128 x 64
  int tm = tile >> 6, tn = tile & 63;
  const __hip_bfloat16* A = xbf + (size_t)tm * 16 * EE;
  const __hip_bfloat16* Bw = wih_bf + (size_t)tn * 16 * EE;
  v8f acc = {};
#pragma unroll
  for (int k0 = 0; k0 < EE; k0 += 32)
    acc = wmma_bf16(load_a_bf16(A, EE, k0, lane), load_b_wt(Bw, EE, k0, lane), acc);
  int col = lane & 15, half = lane >> 4;
  int n = tn * 16 + col;
  float bias = b_ih[n] + b_hh[n];
#pragma unroll
  for (int r = 0; r < 8; ++r) {
    int m = tm * 16 + r + 8 * half;
    xg[(size_t)m * 1024 + n] = acc[r] + bias;
  }
}

// ---------------- LSTM recurrence: one persistent WGP workgroup ----------------
// Per step: gates(8x1024) = h(8x256) @ w_hh^T  via V_WMMA_F32_16X16X4 (h padded to 16 rows),
// then elementwise gate math; masked h stored as bf16 into attended[:, 0:256].
__global__ void __launch_bounds__(1024) lstm_kernel(const float* xg, const float* w_hh,
                                                    const int* lengths,
                                                    __hip_bfloat16* attended) {
  __shared__ float h_lds[16][260];   // rows 8..15 stay zero (pad)
  __shared__ float c_lds[8][256];
  __shared__ float g_lds[8][1024];
  __shared__ int   len_s[8];
  int tid = threadIdx.x;
  for (int i = tid; i < 16 * 260; i += 1024) h_lds[i / 260][i % 260] = 0.f;
  for (int i = tid; i < 8 * 256; i += 1024) c_lds[i >> 8][i & 255] = 0.f;
  if (tid < 8) len_s[tid] = lengths[tid];
  __syncthreads();

  int lane = tid & 31, wave = tid >> 5;
  int l15 = lane & 15, half = lane >> 4;
  int n0a = wave * 32, n0b = wave * 32 + 16;

  for (int t = 0; t < TT; ++t) {
    v8f acc0 = {}, acc1 = {};
#pragma unroll 4
    for (int k0 = 0; k0 < HH; k0 += 4) {
      v2f a;                                   // f32 A 16x4: lanes<16 K={0,1}, >=16 K={2,3}
      a.x = h_lds[l15][k0 + 2 * half];
      a.y = h_lds[l15][k0 + 2 * half + 1];
      v2f b0, b1;                              // B from w_hh^T (rows of w_hh contiguous in K)
      const float* w0 = w_hh + (size_t)(n0a + l15) * HH + k0 + 2 * half;
      const float* w1 = w_hh + (size_t)(n0b + l15) * HH + k0 + 2 * half;
      b0.x = w0[0]; b0.y = w0[1];
      b1.x = w1[0]; b1.y = w1[1];
      acc0 = wmma_f32x4(a, b0, acc0);
      acc1 = wmma_f32x4(a, b1, acc1);
    }
    if (half == 0) {                           // rows 0..7 (=batch) live in lanes 0..15
#pragma unroll
      for (int r = 0; r < 8; ++r) {
        g_lds[r][n0a + l15] = acc0[r] + xg[(size_t)(t * BB + r) * 1024 + n0a + l15];
        g_lds[r][n0b + l15] = acc1[r] + xg[(size_t)(t * BB + r) * 1024 + n0b + l15];
      }
    }
    __syncthreads();
    if (tid < 256) {
      int u = tid;
      for (int bb = 0; bb < BB; ++bb) {
        float ig = g_lds[bb][u], fg = g_lds[bb][256 + u];
        float gg = g_lds[bb][512 + u], og = g_lds[bb][768 + u];
        float c = sigf(fg) * c_lds[bb][u] + sigf(ig) * tanhf(gg);
        float h = sigf(og) * tanhf(c);
        c_lds[bb][u] = c;
        h_lds[bb][u] = h;
        float hm = (t < len_s[bb]) ? h : 0.f;  // enc mask applied only to stored output
        attended[(size_t)(t * BB + bb) * 512 + u] = __float2bfloat16(hm);
      }
    }
    __syncthreads();
  }
}

// ---------------- aq/bk: enc @ [w1^T | w2^T] + [0 | attn_b]  (2048 x 512) ----------------
__global__ void attn_gemm_kernel(const __hip_bfloat16* attended, const __hip_bfloat16* awgt_bf,
                                 const float* attn_b, float* aqbk) {
  int lane = threadIdx.x & 31, wave = threadIdx.x >> 5;
  int tile = blockIdx.x * 4 + wave;           // 4096 tiles: 128 x 32
  int tm = tile >> 5, tn = tile & 31;
  int koff = (tn < 16) ? 0 : HH;              // w1 = attn_w[:, :H], w2 = attn_w[:, H:]
  int nbase = (tn & 15) * 16;
  const __hip_bfloat16* A = attended + (size_t)tm * 16 * 512;  // enc cols 0..255, lda=512
  const __hip_bfloat16* Bw = awgt_bf + (size_t)nbase * (2 * HH) + koff;
  v8f acc = {};
#pragma unroll
  for (int k0 = 0; k0 < HH; k0 += 32)
    acc = wmma_bf16(load_a_bf16(A, 512, k0, lane), load_b_wt(Bw, 2 * HH, k0, lane), acc);
  int col = lane & 15, half = lane >> 4;
  int n = tn * 16 + col;
  float bias = (n >= 256) ? attn_b[n - 256] : 0.f;
#pragma unroll
  for (int r = 0; r < 8; ++r) {
    int m = tm * 16 + r + 8 * half;
    aqbk[(size_t)m * 512 + n] = acc[r] + bias;
  }
}

// ---------------- scores[b,q,k] = sum_h v[h]*tanh(aq[q,b,h]+bk[k,b,h]) ----------------
__global__ void score_kernel(const float* aqbk, const float* v, const int* lengths,
                             float* scores) {
  int q = blockIdx.x, b = blockIdx.y, k = threadIdx.x;
  __shared__ float aqs[HH], vs[HH];
  aqs[k] = aqbk[(size_t)(q * BB + b) * 512 + k];
  vs[k] = v[k];
  __syncthreads();
  float sc = NEGF;
  if (k < lengths[b]) {
    const float* bkrow = aqbk + (size_t)(k * BB + b) * 512 + HH;
    float s = 0.f;
#pragma unroll 4
    for (int h = 0; h < HH; ++h) s += vs[h] * tanhf(aqs[h] + bkrow[h]);
    sc = s;
  }
  scores[(size_t)(b * TT + q) * TT + k] = sc;
}

// ---------------- softmax over k (masked rows already at NEG) -> bf16 probs ----------------
__global__ void softmax_kernel(const float* scores, __hip_bfloat16* aprob_bf) {
  int q = blockIdx.x, b = blockIdx.y, k = threadIdx.x;
  __shared__ float red[TT];
  float val = scores[(size_t)(b * TT + q) * TT + k];
  red[k] = val;
  __syncthreads();
  for (int s2 = 128; s2 > 0; s2 >>= 1) {
    if (k < s2) red[k] = fmaxf(red[k], red[k + s2]);
    __syncthreads();
  }
  float m = red[0];
  __syncthreads();
  float e = expf(val - m);
  red[k] = e;
  __syncthreads();
  for (int s2 = 128; s2 > 0; s2 >>= 1) {
    if (k < s2) red[k] += red[k + s2];
    __syncthreads();
  }
  float inv = 1.f / red[0];
  aprob_bf[(size_t)(b * TT + q) * TT + k] = __float2bfloat16(e * inv);
}

// ---------------- context[q,b,:] = attnw[b,q,:] @ enc[:,b,:]  -> attended cols 256..511 ----
__global__ void context_gemm_kernel(const __hip_bfloat16* aprob_bf, __hip_bfloat16* attended) {
  int lane = threadIdx.x & 31, wave = threadIdx.x >> 5;
  int tile = blockIdx.x * 4 + wave;           // 2048 tiles: b(8) x tq(16) x th(16)
  int b = tile >> 8;
  int rem = tile & 255;
  int tq = rem >> 4, th = rem & 15;
  const __hip_bfloat16* A = aprob_bf + (size_t)b * TT * TT + (size_t)tq * 16 * TT;
  const __hip_bfloat16* Bp = attended + (size_t)b * 512 + th * 16;   // (k,col): +k*4096+col
  v8f acc = {};
#pragma unroll
  for (int k0 = 0; k0 < TT; k0 += 32)
    acc = wmma_bf16(load_a_bf16(A, TT, k0, lane), load_b_strided(Bp, BB * 512, k0, lane), acc);
  int col = lane & 15, half = lane >> 4;
#pragma unroll
  for (int r = 0; r < 8; ++r) {
    int q = tq * 16 + r + 8 * half;
    attended[(size_t)(q * BB + b) * 512 + 256 + th * 16 + col] = __float2bfloat16(acc[r]);
  }
}

// ---------------- projection + fused online log-softmax + gold gather ----------------
// One block per 16 rows; the 16x512 bf16 A-tile is DMA'd to LDS by the Tensor Data Mover
// (one issue from wave 0, tracked by TENSORcnt); 8 waves sweep 2000 column tiles of V
// (proj_w bf16 = 32 MB -> L2-resident) keeping flash-style (m,s) stats per row plus the
// gold logit; final reduce -> atomicAdd into out[b].
__global__ void __launch_bounds__(256) proj_lse_kernel(const __hip_bfloat16* attended,
                                                       const __hip_bfloat16* proj_bf,
                                                       const int* target, float* out) {
  __shared__ __hip_bfloat16 a_lds[16 * 512];
  __shared__ float red_m[8][16], red_s[8][16], red_g[8][16];
  int tid = threadIdx.x;
  int rowbase = blockIdx.x * 16;
#if HAVE_TDM
  if (tid < 32) {
    unsigned lds_off =
        (unsigned)(unsigned long long)(__attribute__((address_space(3))) const void*)&a_lds[0];
    tdm_load_bf16_tile(lds_off,
                       (unsigned long long)(const void*)(attended + (size_t)rowbase * 512),
                       /*cols=*/512u, /*rows=*/16u);
    __builtin_amdgcn_s_wait_tensorcnt(0);
  }
#else
  for (int i = tid; i < 16 * 512; i += 256)
    a_lds[i] = attended[(size_t)(rowbase + (i >> 9)) * 512 + (i & 511)];
#endif
  __syncthreads();

  int lane = tid & 31, wave = tid >> 5;
  int col = lane & 15, half = lane >> 4;
  int tgt[8];
#pragma unroll
  for (int r = 0; r < 8; ++r) tgt[r] = target[rowbase + r + 8 * half];
  float m[8], s[8], g[8];
#pragma unroll
  for (int r = 0; r < 8; ++r) { m[r] = -3.0e38f; s[r] = 0.f; g[r] = 0.f; }

  for (int j = 0; j < 250; ++j) {             // 2000 tiles / 8 waves
    int n0 = (wave + j * 8) * 16;
    // look-ahead prefetch of the next B panel for this wave (L2-resident proj_w)
    __builtin_prefetch(proj_bf + (size_t)(n0 + 128) * 512, 0, 1);
    v8f acc = {};
#pragma unroll
    for (int k0 = 0; k0 < 512; k0 += 32) {
      v16bf a = load_a_bf16(a_lds, 512, k0, lane);
      v16bf b = load_b_wt(proj_bf + (size_t)n0 * 512, 512, k0, lane);
      acc = wmma_bf16(a, b, acc);
    }
    int n = n0 + col;
#pragma unroll
    for (int r = 0; r < 8; ++r) {
      float x = acc[r];
      float nm = fmaxf(m[r], x);
      s[r] = s[r] * expf(m[r] - nm) + expf(x - nm);
      m[r] = nm;
      if (n == tgt[r]) g[r] += x;
    }
  }
  // reduce (m,s,g) across the 16 lanes holding the same rows
  for (int mask = 1; mask < 16; mask <<= 1) {
#pragma unroll
    for (int r = 0; r < 8; ++r) {
      float om = __shfl_xor(m[r], mask, 32);
      float os = __shfl_xor(s[r], mask, 32);
      float og = __shfl_xor(g[r], mask, 32);
      float nm = fmaxf(m[r], om);
      s[r] = s[r] * expf(m[r] - nm) + os * expf(om - nm);
      m[r] = nm;
      g[r] += og;
    }
  }
  if (col == 0) {
#pragma unroll
    for (int r = 0; r < 8; ++r) {
      red_m[wave][r + 8 * half] = m[r];
      red_s[wave][r + 8 * half] = s[r];
      red_g[wave][r + 8 * half] = g[r];
    }
  }
  __syncthreads();
  if (tid < 16) {
    float mm = red_m[0][tid], ss = red_s[0][tid], gg = red_g[0][tid];
    for (int w = 1; w < 8; ++w) {
      float om = red_m[w][tid], os = red_s[w][tid];
      float nm = fmaxf(mm, om);
      ss = ss * expf(mm - nm) + os * expf(om - nm);
      mm = nm;
      gg += red_g[w][tid];
    }
    int grow = rowbase + tid;                 // grow = t*8 + b, flat (T,B)
    int b = grow & 7;
    if (target[grow] != 0)                    // tmask
      atomicAdd(&out[b], gg - (mm + logf(ss)));
  }
}

// ---------------- host: workspace carving + launches ----------------
extern "C" void kernel_launch(void* const* d_in, const int* in_sizes, int n_in,
                              void* d_out, int out_size, void* d_ws, size_t ws_size,
                              hipStream_t stream) {
  (void)in_sizes; (void)n_in; (void)out_size; (void)ws_size;
  const int*   source  = (const int*)d_in[0];
  const int*   target  = (const int*)d_in[1];
  const int*   lengths = (const int*)d_in[2];
  const float* emb     = (const float*)d_in[3];
  const float* w_ih    = (const float*)d_in[4];
  const float* w_hh    = (const float*)d_in[5];
  const float* b_ih    = (const float*)d_in[6];
  const float* b_hh    = (const float*)d_in[7];
  const float* attn_w  = (const float*)d_in[8];
  const float* attn_b  = (const float*)d_in[9];
  const float* v       = (const float*)d_in[10];
  const float* proj_w  = (const float*)d_in[11];
  float* out = (float*)d_out;
  char*  ws  = (char*)d_ws;

  size_t off = 0;
  __hip_bfloat16* x_bf    = (__hip_bfloat16*)(ws + off); off += (size_t)2048 * 256 * 2;
  __hip_bfloat16* wih_bf  = (__hip_bfloat16*)(ws + off); off += (size_t)1024 * 256 * 2;
  __hip_bfloat16* awgt_bf = (__hip_bfloat16*)(ws + off); off += (size_t)256 * 512 * 2;
  __hip_bfloat16* proj_bf = (__hip_bfloat16*)(ws + off); off += (size_t)VV * 512 * 2;
  float*          xg      = (float*)(ws + off);          off += (size_t)2048 * 1024 * 4;
  float*          aqbk    = (float*)(ws + off);          off += (size_t)2048 * 512 * 4;
  float*          scores  = (float*)(ws + off);          off += (size_t)8 * 256 * 256 * 4;
  __hip_bfloat16* aprob   = (__hip_bfloat16*)(ws + off); off += (size_t)8 * 256 * 256 * 2;
  __hip_bfloat16* attended= (__hip_bfloat16*)(ws + off); off += (size_t)2048 * 512 * 2;

  zero_kernel<<<1, 32, 0, stream>>>(out, BB);
  embed_kernel<<<2048, 256, 0, stream>>>(source, emb, x_bf);
  cast_bf16_kernel<<<1024, 256, 0, stream>>>(w_ih, wih_bf, 1024 * 256);
  cast_bf16_kernel<<<512, 256, 0, stream>>>(attn_w, awgt_bf, 256 * 512);
  cast_bf16_kernel<<<64000, 256, 0, stream>>>(proj_w, proj_bf, VV * 512);
  xw_gemm_kernel<<<2048, 128, 0, stream>>>(x_bf, wih_bf, b_ih, b_hh, xg);
  lstm_kernel<<<1, 1024, 0, stream>>>(xg, w_hh, lengths, attended);
  attn_gemm_kernel<<<1024, 128, 0, stream>>>(attended, awgt_bf, attn_b, aqbk);
  score_kernel<<<dim3(256, 8), 256, 0, stream>>>(aqbk, v, lengths, scores);
  softmax_kernel<<<dim3(256, 8), 256, 0, stream>>>(scores, aprob);
  context_gemm_kernel<<<512, 128, 0, stream>>>(aprob, attended);
  proj_lse_kernel<<<128, 256, 0, stream>>>(attended, proj_bf, target, out);
}
